// Mambablock_66907000537452
// MI455X (gfx1250) — matrix operand order repaired and checked by reading, not compiled, not run on hardware
//
#include <hip/hip_runtime.h>
#include <stdint.h>

// ---------------------------------------------------------------------------
// Mamba block for MI455X (gfx1250): bf16 WMMA GEMMs + fused scan.
// Shapes: B=4, L=2048, D=1024, ED=2048, N=16, K=15, R=64, M=B*L=8192.
// ---------------------------------------------------------------------------

typedef __attribute__((ext_vector_type(16))) __bf16 v16bf;
typedef __attribute__((ext_vector_type(8)))  float  v8f;

namespace {
constexpr int cB  = 4;
constexpr int cL  = 2048;
constexpr int cD  = 1024;
constexpr int cED = 2048;
constexpr int cN  = 16;
constexpr int cK  = 15;
constexpr int cR  = 64;
constexpr int cM  = cB * cL;          // 8192 token rows
constexpr int cXZ = 2 * cED;          // 4096
constexpr int cXD = cR + 2 * cN;      // 96
}

// float -> bf16 with round-to-nearest-even
__device__ __forceinline__ uint16_t f2bf(float f) {
    uint32_t u = __float_as_uint(f);
    uint32_t r = u + 0x7fffu + ((u >> 16) & 1u);
    return (uint16_t)(r >> 16);
}

__global__ void cvt_f32_to_bf16(const float* __restrict__ in,
                                uint16_t* __restrict__ out, int n) {
    int i = blockIdx.x * 256 + threadIdx.x;
    if (i < n) out[i] = f2bf(in[i]);
}

// ---------------------------------------------------------------------------
// LayerNorm over D=1024, output bf16 (feeds WMMA GEMM1). One block per row.
// ---------------------------------------------------------------------------
__global__ void ln_to_bf16(const float* __restrict__ x,
                           const float* __restrict__ gamma,
                           const float* __restrict__ beta,
                           uint16_t* __restrict__ xn) {
    __shared__ float s1[256], s2[256];
    const int row = blockIdx.x;
    const float* xr = x + (size_t)row * cD;
    float a = 0.f, b = 0.f;
    for (int i = threadIdx.x; i < cD; i += 256) {
        float v = xr[i];
        a += v; b += v * v;
    }
    s1[threadIdx.x] = a; s2[threadIdx.x] = b;
    __syncthreads();
    for (int off = 128; off > 0; off >>= 1) {
        if ((int)threadIdx.x < off) {
            s1[threadIdx.x] += s1[threadIdx.x + off];
            s2[threadIdx.x] += s2[threadIdx.x + off];
        }
        __syncthreads();
    }
    const float mean = s1[0] * (1.f / cD);
    const float var  = s2[0] * (1.f / cD) - mean * mean;
    const float rstd = rsqrtf(var + 1e-5f);
    for (int i = threadIdx.x; i < cD; i += 256) {
        float v = (xr[i] - mean) * rstd * gamma[i] + beta[i];
        xn[(size_t)row * cD + i] = f2bf(v);
    }
}

// ---------------------------------------------------------------------------
// bf16 WMMA GEMM: Cf/Cb[M,N] = A[M,K] * W[N,K]^T (+ bias[N]) (+ residual).
// Wave computes a 32x32 tile as 2x2 v_wmma_f32_16x16x32_bf16 frags.
// Fragment lane layouts follow CDNA5 ISA §7.12.2:
//   A (16x32, MxK): lane m=lane&15; K-halves 0-7/16-23 (lanes 0-15),
//                   8-15/24-31 (lanes 16-31)  -> two b128 loads per frag.
//   B (32x16, KxN): lane n=lane&15; K 0-7/16-23 (lanes 0-15), 8-15/24-31 (lanes 16-31).
// ---------------------------------------------------------------------------
template <int WM, int WN, bool BIAS, bool RES, bool WF32, bool WBF16>
__global__ __launch_bounds__(WM * WN * 32)
void gemm_bf16_wmma(const uint16_t* __restrict__ A,
                    const uint16_t* __restrict__ W,
                    const float* __restrict__ bias,
                    const float* __restrict__ res,
                    float* __restrict__ Cf,
                    uint16_t* __restrict__ Cb,
                    int K, int lda, int ldw, int ldc) {
    const int lane = threadIdx.x & 31;
    const int wave = threadIdx.x >> 5;
    const int row0 = (blockIdx.y * WM + (wave % WM)) * 32;
    const int col0 = (blockIdx.x * WN + (wave / WM)) * 32;

    v8f acc00 = {}, acc01 = {}, acc10 = {}, acc11 = {};

    const int am = lane & 15;
    const int ak = (lane >> 4) * 8;          // K sub-offset for A frag
    const int bn = lane & 15;
    const int bk = (lane >> 4) * 8;          // K sub-offset for B frag

    const uint16_t* Ap0 = A + (size_t)(row0 + am) * lda + ak;
    const uint16_t* Ap1 = Ap0 + (size_t)16 * lda;
    const uint16_t* Wp0 = W + (size_t)(col0 + bn) * ldw + bk;
    const uint16_t* Wp1 = Wp0 + (size_t)16 * ldw;

    for (int k0 = 0; k0 < K; k0 += 32) {
        union { uint4 u[2]; v16bf v; } a0, a1, b0, b1;
        a0.u[0] = *(const uint4*)(Ap0 + k0);
        a0.u[1] = *(const uint4*)(Ap0 + k0 + 16);
        a1.u[0] = *(const uint4*)(Ap1 + k0);
        a1.u[1] = *(const uint4*)(Ap1 + k0 + 16);
        b0.u[0] = *(const uint4*)(Wp0 + k0);
        b0.u[1] = *(const uint4*)(Wp0 + k0 + 16);
        b1.u[0] = *(const uint4*)(Wp1 + k0);
        b1.u[1] = *(const uint4*)(Wp1 + k0 + 16);

        // Stream-ahead prefetch of the A operand (weights are L2-resident).
        if (k0 + 128 < K) __builtin_prefetch(Ap0 + k0 + 128, 0, 0);

        acc00 = __builtin_amdgcn_wmma_f32_16x16x32_bf16(
            false, a0.v, false, b0.v, (short)0, acc00, false, false);
        acc01 = __builtin_amdgcn_wmma_f32_16x16x32_bf16(
            false, a0.v, false, b1.v, (short)0, acc01, false, false);
        acc10 = __builtin_amdgcn_wmma_f32_16x16x32_bf16(
            false, a1.v, false, b0.v, (short)0, acc10, false, false);
        acc11 = __builtin_amdgcn_wmma_f32_16x16x32_bf16(
            false, a1.v, false, b1.v, (short)0, acc11, false, false);
    }

    // C/D layout (§7.12.2): VGPR r holds M=r (lanes 0-15) / M=8+r (lanes 16-31).
    const int cn = lane & 15;
    const int rb = (lane >> 4) * 8;

    auto store_tile = [&](v8f& acc, int roff, int coff) {
        const int col = col0 + coff + cn;
        const float bv = BIAS ? bias[col] : 0.f;
#pragma unroll
        for (int r = 0; r < 8; ++r) {
            const int row = row0 + roff + rb + r;
            float v = acc[r] + bv;
            if (RES)   v += res[(size_t)row * ldc + col];
            if (WF32)  Cf[(size_t)row * ldc + col] = v;
            if (WBF16) Cb[(size_t)row * ldc + col] = f2bf(v);
        }
    };
    store_tile(acc00, 0, 0);
    store_tile(acc01, 0, 16);
    store_tile(acc10, 16, 0);
    store_tile(acc11, 16, 16);
}

// ---------------------------------------------------------------------------
// Depthwise causal conv (K=15) + bias + SiLU. Reads xin = xz[:, :ED].
// Writes fp32 (scan) and bf16 (GEMM2) copies. Coalesced along e.
// ---------------------------------------------------------------------------
__global__ void conv_silu(const float* __restrict__ xz,
                          const float* __restrict__ cw,
                          const float* __restrict__ cb,
                          float* __restrict__ xc,
                          uint16_t* __restrict__ xcb) {
    const int e = blockIdx.x * 256 + threadIdx.x;
    const int l = blockIdx.y;
    const int b = blockIdx.z;
    float acc = cb[e];
#pragma unroll
    for (int k = 0; k < cK; ++k) {
        const int ll = l - (cK - 1) + k;
        if (ll >= 0)
            acc += xz[(size_t)(b * cL + ll) * cXZ + e] * cw[e * cK + k];
    }
    const float s = acc / (1.f + __expf(-acc));
    const size_t o = (size_t)(b * cL + l) * cED + e;
    xc[o]  = s;
    xcb[o] = f2bf(s);
}

// ---------------------------------------------------------------------------
// Selective scan: one thread per (b, e) channel, N=16 states in registers.
// B_t/C_t (32 floats) broadcast via LDS per timestep. Fuses softplus(delta),
// the recurrence, D-skip, and SiLU(z) gating; emits bf16 y for GEMM4.
// ---------------------------------------------------------------------------
__global__ __launch_bounds__(256)
void ssm_scan(const float* __restrict__ delta_raw,
              const float* __restrict__ xdbl,
              const float* __restrict__ xc,
              const float* __restrict__ xz,
              const float* __restrict__ A_log,
              const float* __restrict__ Dp,
              uint16_t* __restrict__ y2b) {
    __shared__ float sBC[32];
    const int e = blockIdx.x * 256 + threadIdx.x;
    const int b = blockIdx.y;

    float Ac[cN], h[cN];
#pragma unroll
    for (int n = 0; n < cN; ++n) {
        Ac[n] = -__expf(A_log[(size_t)e * cN + n]);
        h[n]  = 0.f;
    }
    const float dpe = Dp[e];

    for (int l = 0; l < cL; ++l) {
        const size_t row = (size_t)(b * cL + l);
        if (threadIdx.x < 32)
            sBC[threadIdx.x] = xdbl[row * cXD + cR + threadIdx.x];
        __syncthreads();

        const float draw = delta_raw[row * cED + e];
        const float dt = (draw > 20.f) ? draw : log1pf(__expf(draw));
        const float xt = xc[row * cED + e];
        float y = 0.f;
#pragma unroll
        for (int n = 0; n < cN; ++n) {
            const float dA = __expf(dt * Ac[n]);
            h[n] = h[n] * dA + dt * sBC[n] * xt;
            y += h[n] * sBC[cN + n];
        }
        const float zz = xz[row * cXZ + cED + e];
        const float yv = (y + xt * dpe) * (zz / (1.f + __expf(-zz)));
        y2b[row * cED + e] = f2bf(yv);
        __syncthreads();
    }
}

// ---------------------------------------------------------------------------
extern "C" void kernel_launch(void* const* d_in, const int* in_sizes, int n_in,
                              void* d_out, int out_size, void* d_ws, size_t ws_size,
                              hipStream_t stream) {
    (void)in_sizes; (void)n_in; (void)out_size; (void)ws_size;

    const float* x       = (const float*)d_in[0];
    const float* gamma   = (const float*)d_in[1];
    const float* beta    = (const float*)d_in[2];
    const float* W_in    = (const float*)d_in[3];
    const float* b_in    = (const float*)d_in[4];
    const float* conv_w  = (const float*)d_in[5];
    const float* conv_b  = (const float*)d_in[6];
    const float* W_x     = (const float*)d_in[7];
    const float* W_dt    = (const float*)d_in[8];
    const float* b_dt    = (const float*)d_in[9];
    const float* A_log   = (const float*)d_in[10];
    const float* D_param = (const float*)d_in[11];
    const float* W_out   = (const float*)d_in[12];
    const float* b_out   = (const float*)d_in[13];
    float* out = (float*)d_out;

    // ---- workspace layout (256 B aligned) ----
    char* ws = (char*)d_ws;
    size_t ofs = 0;
    auto alloc = [&](size_t bytes) -> void* {
        void* p = ws + ofs;
        ofs = (ofs + bytes + 255) & ~(size_t)255;
        return p;
    };
    uint16_t* xn_bf   = (uint16_t*)alloc((size_t)cM * cD * 2);        // 16 MB
    float*    xz      = (float*)   alloc((size_t)cM * cXZ * 4);       // 128 MB
    float*    xc      = (float*)   alloc((size_t)cM * cED * 4);       // 64 MB
    uint16_t* xc_bf   = (uint16_t*)alloc((size_t)cM * cED * 2);       // 32 MB
    float*    xdbl    = (float*)   alloc((size_t)cM * cXD * 4);       // 3 MB
    uint16_t* xdbl_bf = (uint16_t*)alloc((size_t)cM * cXD * 2);       // 1.5 MB
    float*    draw    = (float*)   alloc((size_t)cM * cED * 4);       // 64 MB
    uint16_t* y2_bf   = (uint16_t*)alloc((size_t)cM * cED * 2);       // 32 MB
    uint16_t* Win_bf  = (uint16_t*)alloc((size_t)cXZ * cD * 2);       // 8 MB
    uint16_t* Wx_bf   = (uint16_t*)alloc((size_t)cXD * cED * 2);
    uint16_t* Wdt_bf  = (uint16_t*)alloc((size_t)cED * cR * 2);
    uint16_t* Wout_bf = (uint16_t*)alloc((size_t)cD * cED * 2);       // 4 MB

    // ---- weight fp32 -> bf16 ----
    auto cvt = [&](const float* src, uint16_t* dst, int n) {
        cvt_f32_to_bf16<<<(n + 255) / 256, 256, 0, stream>>>(src, dst, n);
    };
    cvt(W_in,  Win_bf,  cXZ * cD);
    cvt(W_x,   Wx_bf,   cXD * cED);
    cvt(W_dt,  Wdt_bf,  cED * cR);
    cvt(W_out, Wout_bf, cD * cED);

    // ---- 1. LayerNorm -> bf16 ----
    ln_to_bf16<<<cM, 256, 0, stream>>>(x, gamma, beta, xn_bf);

    // ---- 2. xz = xn @ W_in^T + b_in   (8192x1024 x 1024x4096) ----
    gemm_bf16_wmma<2, 4, true, false, true, false>
        <<<dim3(cXZ / 128, cM / 64), 256, 0, stream>>>(
            xn_bf, Win_bf, b_in, nullptr, xz, nullptr,
            cD, cD, cD, cXZ);

    // ---- 3. depthwise causal conv + SiLU ----
    conv_silu<<<dim3(cED / 256, cL, cB), 256, 0, stream>>>(
        xz, conv_w, conv_b, xc, xc_bf);

    // ---- 4. xdbl = xc @ W_x^T   (8192x2048 x 2048x96), fp32 + bf16 out ----
    gemm_bf16_wmma<8, 1, false, false, true, true>
        <<<dim3(cXD / 32, cM / 256), 256, 0, stream>>>(
            xc_bf, Wx_bf, nullptr, nullptr, xdbl, xdbl_bf,
            cED, cED, cED, cXD);

    // ---- 5. delta_raw = dt_low @ W_dt^T + b_dt   (8192x64 x 64x2048) ----
    gemm_bf16_wmma<2, 4, true, false, true, false>
        <<<dim3(cED / 128, cM / 64), 256, 0, stream>>>(
            xdbl_bf, Wdt_bf, b_dt, nullptr, draw, nullptr,
            cR, cXD, cR, cED);

    // ---- 6. selective scan + gating -> bf16 y ----
    ssm_scan<<<dim3(cED / 256, cB), 256, 0, stream>>>(
        draw, xdbl, xc, xz, A_log, D_param, y2_bf);

    // ---- 7. out = y @ W_out^T + b_out + x   (8192x2048 x 2048x1024) ----
    gemm_bf16_wmma<2, 4, true, true, true, false>
        <<<dim3(cD / 128, cM / 64), 256, 0, stream>>>(
            y2_bf, Wout_bf, b_out, x, out, nullptr,
            cED, cED, cED, cD);
}
